// Attention_35639638622706
// MI455X (gfx1250) — matrix operand order repaired and checked by reading, not compiled, not run on hardware
//
#include <hip/hip_runtime.h>

// Problem constants (B, S, DIM, H, DH) = (2, 2048, 1024, 16, 64)
#define B_    2
#define S_    2048
#define DIM_  1024
#define H_    16
#define DH_   64
#define BS_   (B_ * S_)      // 4096 rows
#define QKVN_ (3 * DIM_)     // 3072

#define TRY_ASYNC 1
#if defined(__gfx1250__) && defined(TRY_ASYNC) && \
    __has_builtin(__builtin_amdgcn_global_load_async_to_lds_b128) && \
    __has_builtin(__builtin_amdgcn_s_wait_asynccnt)
#define USE_ASYNC 1
#else
#define USE_ASYNC 0
#endif

typedef __attribute__((ext_vector_type(16))) __bf16         v16bf;
typedef __attribute__((ext_vector_type(8)))  float          v8f;
typedef __attribute__((ext_vector_type(8)))  unsigned short v8u;
typedef __attribute__((ext_vector_type(4)))  int            v4i;

union FragBF { v16bf v; v8u u[2]; };
static_assert(sizeof(FragBF) == 32, "frag size");

__device__ __forceinline__ unsigned short f2bf(float f) {
  union { float f; unsigned u; } x; x.f = f;
  unsigned r = x.u + 0x7fffu + ((x.u >> 16) & 1u);   // round-to-nearest-even
  return (unsigned short)(r >> 16);
}

// ---- WMMA fragment loaders (layouts per CDNA5 ISA §7.12.2) ----
// A (16xK, 16-bit): lane L holds row M=L%16; K chunks {kb..kb+7} and
// {16+kb..16+kb+7} with kb = 8*(L/16), within each 32-wide K step.
__device__ __forceinline__ v16bf load_a_frag(const unsigned short* base,
                                             int row, int ld, int k0) {
  const int lane = threadIdx.x & 31;
  const int r    = lane & 15;
  const int kb   = (lane >> 4) << 3;
  const unsigned short* p = base + (size_t)(row + r) * ld + k0;
  FragBF f;
  f.u[0] = *(const v8u*)(p + kb);
  f.u[1] = *(const v8u*)(p + 16 + kb);
  return f.v;
}
// B (Kx16, 16-bit): lane L holds col N=L%16; K chunks {ks..ks+7} and
// {16+ks..16+ks+7} with ks = 8*(L/16). Source is row-major [N][K] (i.e. B^T rows).
__device__ __forceinline__ v16bf load_b_frag(const unsigned short* base,
                                             int row, int ld, int k0) {
  const int lane = threadIdx.x & 31;
  const int r    = lane & 15;
  const int ks   = (lane >> 4) << 3;
  const unsigned short* p = base + (size_t)(row + r) * ld + k0 + ks;
  FragBF f;
  f.u[0] = *(const v8u*)(p);
  f.u[1] = *(const v8u*)(p + 16);
  return f.v;
}

__device__ __forceinline__ v8f wmma_bf16(v16bf a, v16bf b, v8f c) {
  return __builtin_amdgcn_wmma_f32_16x16x32_bf16(false, a, false, b,
                                                 (short)0, c, false, false);
}

#if USE_ASYNC
// Builtin prototype (from hipcc diagnostic): param0 = v4i in AS1 ("__device__"),
// param1 = v4i in AS3, then imm offset + imm cpol.
__device__ __forceinline__ void async_copy16(const unsigned short* g,
                                             unsigned short* l) {
  auto gp = (__attribute__((address_space(1))) v4i*)(unsigned long long)g;
  auto lp = (__attribute__((address_space(3))) v4i*)
            (unsigned int)(unsigned long long)l;   // low 32 bits = LDS offset
  __builtin_amdgcn_global_load_async_to_lds_b128(gp, lp, 0, 0);
}
#endif

// ---------------- kernel 0: f32 -> bf16 conversion ----------------
__global__ void k_cvt_bf16(const float* __restrict__ src,
                           unsigned short* __restrict__ dst, int n) {
  int i = blockIdx.x * blockDim.x + threadIdx.x;
  int stride = gridDim.x * blockDim.x;
  for (; i < n; i += stride) dst[i] = f2bf(src[i]);
}

// ---------------- kernel 1: QKV GEMM + RMSNorm + RoPE -------------
// grid (64, 48), block 128 (4 waves). Wave tile: 16(M) x 64(N) = one head
// slice of q/k/v for 16 sequence positions. N tile index selects q/k/v+head.
__global__ void __launch_bounds__(128)
k_qkv(const unsigned short* __restrict__ xb,     // (BS, DIM) bf16
      const unsigned short* __restrict__ wb,     // (3*DIM, DIM) bf16
      const float* __restrict__ qw, const float* __restrict__ kw,
      const float* __restrict__ freqs,           // (S, DH) f32
      unsigned short* __restrict__ qout,         // (B,H,S,DH) bf16
      unsigned short* __restrict__ kout,         // (B,H,S,DH) bf16
      unsigned short* __restrict__ vtout)        // (B,H,DH,S) bf16 (transposed)
{
  __shared__ __align__(16) unsigned short wst[64 * 32];   // W tile stage
  __shared__ __align__(16) float ep[4][16 * 64];          // epilogue stage

  const int t    = threadIdx.x;
  const int w    = t >> 5;
  const int lane = t & 31;
  const int m0   = blockIdx.x * 64 + w * 16;
  const int n0   = blockIdx.y * 64;

  v8f acc[4] = {};
  const int srow = t >> 1;            // 128 threads stage 64x32 bf16 tile
  const int scol = (t & 1) * 16;
  const unsigned short* wsrc = wb + (size_t)(n0 + srow) * DIM_ + scol;
  unsigned short* wdst = &wst[srow * 32 + scol];

  for (int kk = 0; kk < DIM_; kk += 32) {
#if USE_ASYNC
    async_copy16(wsrc + kk, wdst);
    async_copy16(wsrc + kk + 8, wdst + 8);
    __builtin_amdgcn_s_wait_asynccnt(0);
#else
    *(v8u*)wdst       = *(const v8u*)(wsrc + kk);
    *(v8u*)(wdst + 8) = *(const v8u*)(wsrc + kk + 8);
#endif
    __syncthreads();
    if (kk + 32 < DIM_)
      __builtin_prefetch(xb + (size_t)(m0 + (lane & 15)) * DIM_ + kk + 32, 0, 1);
    v16bf a = load_a_frag(xb, m0, DIM_, kk);
#pragma unroll
    for (int nb = 0; nb < 4; ++nb) {
      v16bf bfr = load_b_frag(wst, nb * 16, 32, 0);
      acc[nb] = wmma_bf16(a, bfr, acc[nb]);
    }
    __syncthreads();
  }

  // ---- epilogue: C tile -> LDS, then per-head RMSNorm + RoPE ----
  float* myep = ep[w];
  const int hi = lane >> 4;
  const int cn = lane & 15;
#pragma unroll
  for (int nb = 0; nb < 4; ++nb)
#pragma unroll
    for (int r = 0; r < 8; ++r)
      myep[(r + 8 * hi) * 64 + nb * 16 + cn] = acc[nb][r];
  __syncthreads();

  const int sec = n0 >> 10;            // 0=q, 1=k, 2=v
  const int h   = (n0 & 1023) >> 6;

  if (sec == 2) {
    // V: store transposed (B,H,DH,S) so PV B-fragments are contiguous.
#pragma unroll 4
    for (int i = 0; i < 32; ++i) {
      int idx = i * 32 + lane;
      int ml = idx >> 6, d = idx & 63;
      int mg = m0 + ml;
      int b = mg >> 11, s = mg & 2047;
      vtout[((size_t)(b * H_ + h) * DH_ + d) * S_ + s] = f2bf(myep[ml * 64 + d]);
    }
  } else {
    const float* nw = (sec == 0) ? qw : kw;
    unsigned short* dst = (sec == 0) ? qout : kout;
    const int ml = lane & 15;
    const int mg = m0 + ml;
    const int b = mg >> 11, s = mg & 2047;
    float ss = 0.f;                      // lane sums its 32-wide half row
#pragma unroll
    for (int j = 0; j < 32; ++j) {
      float v = myep[ml * 64 + hi * 32 + j];
      ss += v * v;
    }
    ss += __shfl_xor(ss, 16, 32);        // lanes L and L^16 share row ml
    const float norm = rsqrtf(ss * (1.0f / 64.0f) + 1e-6f);
    unsigned short* orow = dst + ((size_t)(b * H_ + h) * S_ + s) * DH_;
#pragma unroll
    for (int j = 0; j < 32; ++j) {
      float x1 = myep[ml * 64 + j]      * norm * nw[j];
      float x2 = myep[ml * 64 + 32 + j] * norm * nw[32 + j];
      float fr = freqs[s * DH_ + j];
      float c = __cosf(fr), sn = __sinf(fr);
      float o = hi ? (x2 * c + x1 * sn) : (x1 * c - x2 * sn);
      orow[hi * 32 + j] = f2bf(o);
    }
  }
}

// ---------------- kernel 2: causal flash attention ----------------
// grid (16, 32), block 256 (8 waves). Wave = 16-query tile of one (b,h);
// streams 32-key blocks with online softmax, all matmuls via WMMA bf16.
__global__ void __launch_bounds__(256)
k_attn(const unsigned short* __restrict__ qb,    // (B,H,S,DH)
       const unsigned short* __restrict__ kbm,   // (B,H,S,DH)
       const unsigned short* __restrict__ vtb,   // (B,H,DH,S)
       unsigned short* __restrict__ attb)        // (B,S,H*DH)
{
  __shared__ __align__(16) unsigned short ps[8][16 * 32];  // per-wave P stage

  const int w    = threadIdx.x >> 5;
  const int lane = threadIdx.x & 31;
  const int q0   = (blockIdx.x * 8 + w) * 16;
  const int bh   = blockIdx.y;
  const int b = bh >> 4, h = bh & 15;

  const unsigned short* qbase = qb  + (size_t)bh * S_ * DH_;
  const unsigned short* kbase = kbm + (size_t)bh * S_ * DH_;
  const unsigned short* vbase = vtb + (size_t)bh * DH_ * S_;

  v16bf qa[2];
  qa[0] = load_a_frag(qbase, q0, DH_, 0);
  qa[1] = load_a_frag(qbase, q0, DH_, 32);

  v8f o[4] = {};
  float rmax[8], rsum[8];
#pragma unroll
  for (int r = 0; r < 8; ++r) { rmax[r] = -1e30f; rsum[r] = 0.f; }

  const int hi = lane >> 4;
  const int cn = lane & 15;
  unsigned short* myps = ps[w];

  for (int t0 = 0; t0 <= q0 + 15; t0 += 32) {   // causal key-block range
    // ---- scores S = (Q K^T) * scale ----
    v8f sf[2];
#pragma unroll
    for (int nt = 0; nt < 2; ++nt) {
      v8f s = {};
      s = wmma_bf16(qa[0], load_b_frag(kbase, t0 + nt * 16, DH_, 0),  s);
      s = wmma_bf16(qa[1], load_b_frag(kbase, t0 + nt * 16, DH_, 32), s);
      sf[nt] = s;
    }
    // ---- scale + causal mask + online softmax ----
    float tmax[8];
#pragma unroll
    for (int r = 0; r < 8; ++r) {
      const int m = q0 + r + 8 * hi;
#pragma unroll
      for (int nt = 0; nt < 2; ++nt) {
        const int tcol = t0 + nt * 16 + cn;
        float v = sf[nt][r] * 0.125f;           // DH^-0.5
        if (tcol > m) v = -1e30f;
        sf[nt][r] = v;
      }
      float mx = fmaxf(sf[0][r], sf[1][r]);
#pragma unroll
      for (int msk = 1; msk < 16; msk <<= 1)    // stays within 16-lane halves
        mx = fmaxf(mx, __shfl_xor(mx, msk, 32));
      tmax[r] = mx;
    }
#pragma unroll
    for (int r = 0; r < 8; ++r) {
      const float mnew  = fmaxf(rmax[r], tmax[r]);
      const float alpha = __expf(rmax[r] - mnew);
      float p0 = __expf(sf[0][r] - mnew);
      float p1 = __expf(sf[1][r] - mnew);
      sf[0][r] = p0; sf[1][r] = p1;
      float psum = p0 + p1;
#pragma unroll
      for (int msk = 1; msk < 16; msk <<= 1)
        psum += __shfl_xor(psum, msk, 32);
      rsum[r] = rsum[r] * alpha + psum;
      rmax[r] = mnew;
#pragma unroll
      for (int nb = 0; nb < 4; ++nb) o[nb][r] *= alpha;
    }
    // ---- P: C-layout -> A-layout transpose through LDS (bf16) ----
#pragma unroll
    for (int r = 0; r < 8; ++r) {
      const int ml = r + 8 * hi;
      myps[ml * 32 + cn]      = f2bf(sf[0][r]);
      myps[ml * 32 + 16 + cn] = f2bf(sf[1][r]);
    }
    __builtin_amdgcn_fence(__ATOMIC_ACQ_REL, "wavefront");
    __builtin_amdgcn_wave_barrier();
    v16bf pa = load_a_frag(myps, 0, 32, 0);
    // ---- O += P V  (V^T layout makes B-fragments contiguous) ----
#pragma unroll
    for (int nb = 0; nb < 4; ++nb)
      o[nb] = wmma_bf16(pa, load_b_frag(vbase, nb * 16, S_, t0), o[nb]);
    __builtin_amdgcn_fence(__ATOMIC_ACQ_REL, "wavefront");
    __builtin_amdgcn_wave_barrier();
  }
  // ---- normalize and store (B, S, H*DH) bf16 for the output GEMM ----
#pragma unroll
  for (int r = 0; r < 8; ++r) {
    const float inv = 1.0f / rsum[r];
    const int m = q0 + r + 8 * hi;
    unsigned short* orow = attb + (size_t)(b * S_ + m) * (H_ * DH_) + h * DH_;
#pragma unroll
    for (int nb = 0; nb < 4; ++nb)
      orow[nb * 16 + cn] = f2bf(o[nb][r] * inv);
  }
}

// ---------------- kernel 3: output projection GEMM ----------------
// grid (64, 16), block 128. out[4096,1024] f32 = attn[4096,1024] @ w_out^T
__global__ void __launch_bounds__(128)
k_out(const unsigned short* __restrict__ ab,     // (BS, DIM) bf16
      const unsigned short* __restrict__ wob,    // (DIM, DIM) bf16
      float* __restrict__ out)
{
  __shared__ __align__(16) unsigned short wst[64 * 32];

  const int t    = threadIdx.x;
  const int w    = t >> 5;
  const int lane = t & 31;
  const int m0   = blockIdx.x * 64 + w * 16;
  const int n0   = blockIdx.y * 64;

  v8f acc[4] = {};
  const int srow = t >> 1;
  const int scol = (t & 1) * 16;
  const unsigned short* wsrc = wob + (size_t)(n0 + srow) * DIM_ + scol;
  unsigned short* wdst = &wst[srow * 32 + scol];

  for (int kk = 0; kk < DIM_; kk += 32) {
#if USE_ASYNC
    async_copy16(wsrc + kk, wdst);
    async_copy16(wsrc + kk + 8, wdst + 8);
    __builtin_amdgcn_s_wait_asynccnt(0);
#else
    *(v8u*)wdst       = *(const v8u*)(wsrc + kk);
    *(v8u*)(wdst + 8) = *(const v8u*)(wsrc + kk + 8);
#endif
    __syncthreads();
    if (kk + 32 < DIM_)
      __builtin_prefetch(ab + (size_t)(m0 + (lane & 15)) * DIM_ + kk + 32, 0, 1);
    v16bf a = load_a_frag(ab, m0, DIM_, kk);
#pragma unroll
    for (int nb = 0; nb < 4; ++nb) {
      v16bf bfr = load_b_frag(wst, nb * 16, 32, 0);
      acc[nb] = wmma_bf16(a, bfr, acc[nb]);
    }
    __syncthreads();
  }
  const int hi = lane >> 4, cn = lane & 15;
#pragma unroll
  for (int nb = 0; nb < 4; ++nb)
#pragma unroll
    for (int r = 0; r < 8; ++r)
      out[(size_t)(m0 + r + 8 * hi) * DIM_ + n0 + nb * 16 + cn] = acc[nb][r];
}

extern "C" void kernel_launch(void* const* d_in, const int* in_sizes, int n_in,
                              void* d_out, int out_size, void* d_ws, size_t ws_size,
                              hipStream_t stream) {
  (void)in_sizes; (void)n_in; (void)out_size; (void)ws_size;
  const float* x     = (const float*)d_in[0];
  // d_in[1] = mask: unused (causal mask applied analytically)
  const float* freqs = (const float*)d_in[2];
  const float* w_qkv = (const float*)d_in[3];
  const float* w_out = (const float*)d_in[4];
  const float* qnw   = (const float*)d_in[5];
  const float* knw   = (const float*)d_in[6];
  float* out = (float*)d_out;

  // Workspace layout (bf16 u16), total ~48 MiB
  unsigned short* ws    = (unsigned short*)d_ws;
  unsigned short* xb    = ws;                                    //  8 MiB
  unsigned short* wqkvb = xb    + (size_t)BS_ * DIM_;            //  6 MiB
  unsigned short* woutb = wqkvb + (size_t)QKVN_ * DIM_;          //  2 MiB
  unsigned short* qb    = woutb + (size_t)DIM_ * DIM_;           //  8 MiB
  unsigned short* kb    = qb    + (size_t)B_ * H_ * S_ * DH_;    //  8 MiB
  unsigned short* vtb   = kb    + (size_t)B_ * H_ * S_ * DH_;    //  8 MiB
  unsigned short* attb  = vtb   + (size_t)B_ * H_ * S_ * DH_;    //  8 MiB

  k_cvt_bf16<<<256, 256, 0, stream>>>(x,     xb,    BS_ * DIM_);
  k_cvt_bf16<<<256, 256, 0, stream>>>(w_qkv, wqkvb, QKVN_ * DIM_);
  k_cvt_bf16<<<256, 256, 0, stream>>>(w_out, woutb, DIM_ * DIM_);
  k_qkv <<<dim3(64, 48), 128, 0, stream>>>(xb, wqkvb, qnw, knw, freqs, qb, kb, vtb);
  k_attn<<<dim3(16, 32), 256, 0, stream>>>(qb, kb, vtb, attb);
  k_out <<<dim3(64, 16), 128, 0, stream>>>(attb, woutb, out);
}
